// Attention_53317724013188
// MI455X (gfx1250) — matrix operand order repaired and checked by reading, not compiled, not run on hardware
//
#include <hip/hip_runtime.h>
#include <hip/hip_bf16.h>

typedef __attribute__((ext_vector_type(16))) _Float16 v16h;
typedef __attribute__((ext_vector_type(8)))  _Float16 v8h;
typedef __attribute__((ext_vector_type(4)))  _Float16 v4h;
typedef __attribute__((ext_vector_type(8)))  float    v8f;
typedef _Float16 h16;

constexpr int BB = 2, TT = 1024, CC = 1024, HH = 16, HD = 64;

// ---------------------------------------------------------------------------
// WMMA helpers (CDNA5 wave32, V_WMMA_F32_16X16X32_F16)
// ---------------------------------------------------------------------------
__device__ __forceinline__ v8f wmma32(v16h a, v16h b, v8f c) {
  return __builtin_amdgcn_wmma_f32_16x16x32_f16(false, a, false, b, (short)0, c,
                                                false, false);
}

// A fragment: 16x32 f16 tile, row-major, row stride lda (halfs, multiple of 8).
// Lane layout (ISA 7.12.2): M = lane&15; lanes 0-15 K={0..7,16..23},
// lanes 16-31 K={8..15,24..31}.  Both K runs are 8 contiguous halfs ->
// exactly two ds_load_b128 per lane.
__device__ __forceinline__ v16h frag_a(const h16* A, int lda, int lane) {
  const int row = lane & 15, hs = lane >> 4;
  const h16* p = A + row * lda + hs * 8;
  v8h lo = *(const v8h*)(p);
  v8h hi = *(const v8h*)(p + 16);
  return __builtin_shufflevector(lo, hi, 0, 1, 2, 3, 4, 5, 6, 7,
                                 8, 9, 10, 11, 12, 13, 14, 15);
}

// B fragment from a TRANSPOSED tile Bt[n][k] (16 rows x 32 k), stride ldb.
// Lane n=lane&15 needs K = {hs*8..hs*8+7, 16+hs*8..16+hs*8+7} (same as A) ->
// two ds_load_b128 per lane.
__device__ __forceinline__ v16h frag_bt(const h16* Bt, int ldb, int lane) {
  const int col = lane & 15, hs = lane >> 4;
  const h16* p = Bt + col * ldb + hs * 8;
  v8h lo = *(const v8h*)(p);
  v8h hi = *(const v8h*)(p + 16);
  return __builtin_shufflevector(lo, hi, 0, 1, 2, 3, 4, 5, 6, 7,
                                 8, 9, 10, 11, 12, 13, 14, 15);
}

// ---------------------------------------------------------------------------
// Linear layer: out = X @ W^T + bias.  X:[M,K] f32 or f16, W:[N,K] f32.
// Block tile 128x64, 128 threads = 4 waves, each wave a 32x64 strip
// (2 A-frags x 4 B-frags = 8 WMMAs per K-step).  W is staged untransposed
// (its [N,K] layout is already the Bt fragment layout).
// ---------------------------------------------------------------------------
template <typename TIN, bool HEAD_MAJOR>
__global__ void __launch_bounds__(128)
linear_kernel(const TIN* __restrict__ X, const float* __restrict__ W,
              const float* __restrict__ bias, void* __restrict__ outv,
              int M, int N, int K) {
  constexpr int BM = 128, BN = 64, BK = 32;
  constexpr int LDA = BK + 8;  // 40 halfs = 80B (16B multiple)
  constexpr int LDB = BK + 8;
  __shared__ __align__(16) h16 sA[BM * LDA];
  __shared__ __align__(16) h16 sBt[BN * LDB];

  const int tid = threadIdx.x, lane = tid & 31, wv = tid >> 5;
  const int m0 = blockIdx.y * BM, n0 = blockIdx.x * BN;

  v8f acc[2][4] = {};

  for (int k0 = 0; k0 < K; k0 += BK) {
    // ---- stage A tile [BM x BK] (vectorized, coalesced) ----
    if constexpr (sizeof(TIN) == 4) {
      for (int idx = tid; idx < BM * (BK / 4); idx += 128) {
        int r = idx >> 3, c = (idx & 7) * 4;
        const float4 f = *(const float4*)&X[(size_t)(m0 + r) * K + k0 + c];
        v4h h = {(h16)f.x, (h16)f.y, (h16)f.z, (h16)f.w};
        *(v4h*)&sA[r * LDA + c] = h;
      }
    } else {
      for (int idx = tid; idx < BM * (BK / 8); idx += 128) {
        int r = idx >> 2, c = (idx & 3) * 8;
        *(v8h*)&sA[r * LDA + c] = *(const v8h*)&X[(size_t)(m0 + r) * K + k0 + c];
      }
    }
    // ---- stage Bt tile [BN x BK] = W rows, no transpose needed ----
    for (int idx = tid; idx < BN * (BK / 4); idx += 128) {
      int n = idx >> 3, c = (idx & 7) * 4;
      const float4 f = *(const float4*)&W[(size_t)(n0 + n) * K + k0 + c];
      v4h h = {(h16)f.x, (h16)f.y, (h16)f.z, (h16)f.w};
      *(v4h*)&sBt[n * LDB + c] = h;
    }
    if (k0 + BK < K) {  // hint next K-tile into cache (global_prefetch_b8)
      __builtin_prefetch(&X[(size_t)(m0 + tid) * K + k0 + BK], 0, 0);
      if (tid < BN) __builtin_prefetch(&W[(size_t)(n0 + tid) * K + k0 + BK], 0, 0);
    }
    __syncthreads();

    v16h a0 = frag_a(&sA[(wv * 32) * LDA], LDA, lane);
    v16h a1 = frag_a(&sA[(wv * 32 + 16) * LDA], LDA, lane);
#pragma unroll
    for (int j = 0; j < 4; ++j) {
      v16h b = frag_bt(&sBt[(j * 16) * LDB], LDB, lane);
      acc[0][j] = wmma32(a0, b, acc[0][j]);
      acc[1][j] = wmma32(a1, b, acc[1][j]);
    }
    __syncthreads();
  }

  // C/D layout: VGPR r -> row hs*8+r of each 16-row tile, N = lane&15.
  const int c0 = lane & 15, hs = lane >> 4;
#pragma unroll
  for (int i = 0; i < 2; ++i) {
#pragma unroll
    for (int j = 0; j < 4; ++j) {
#pragma unroll
      for (int r = 0; r < 8; ++r) {
        int gm = m0 + wv * 32 + i * 16 + hs * 8 + r;
        int gn = n0 + j * 16 + c0;
        float val = acc[i][j][r] + bias[gn];
        if constexpr (HEAD_MAJOR) {
          int b = gm / TT, t = gm % TT;
          int h = gn / HD, d = gn % HD;
          ((h16*)outv)[(((size_t)b * HH + h) * TT + t) * HD + d] = (h16)val;
        } else {
          ((float*)outv)[(size_t)gm * N + gn] = val;
        }
      }
    }
  }
}

// ---------------------------------------------------------------------------
// Scores: S[bh][t][s] = q.k / sqrt(HD), causal -inf mask.
// 64x64 tile per block; K-operand tile staged untransposed (k's [t][d] rows
// are already the Bt layout since N=t, K=d).
// ---------------------------------------------------------------------------
__global__ void __launch_bounds__(128)
scores_kernel(const h16* __restrict__ q, const h16* __restrict__ k,
              float* __restrict__ S) {
  constexpr int LDQ = HD + 8;  // 72 halfs = 144B (16B multiple)
  constexpr int LDK = HD + 8;
  __shared__ __align__(16) h16 sQ[64 * LDQ];   // [t][d]
  __shared__ __align__(16) h16 sKt[64 * LDK];  // [s][d]  == Bt layout

  const int tid = threadIdx.x, lane = tid & 31, wv = tid >> 5;
  const int bh = blockIdx.z;
  const int row0 = blockIdx.y * 64, col0 = blockIdx.x * 64;
  float* Sp = S + (size_t)bh * TT * TT;

  if (col0 > row0 + 63) {  // fully above diagonal: -inf fill, no WMMA work
    const float4 ninf = {-__builtin_inff(), -__builtin_inff(),
                         -__builtin_inff(), -__builtin_inff()};
    for (int idx = tid; idx < 64 * 16; idx += 128) {
      int r = idx >> 4, c = (idx & 15) * 4;
      *(float4*)&Sp[(size_t)(row0 + r) * TT + col0 + c] = ninf;
    }
    return;
  }

  const h16* qp = q + (size_t)bh * TT * HD;
  const h16* kp = k + (size_t)bh * TT * HD;
  for (int idx = tid; idx < 64 * (HD / 8); idx += 128) {
    int r = idx >> 3, c = (idx & 7) * 8;
    *(v8h*)&sQ[r * LDQ + c]  = *(const v8h*)&qp[(size_t)(row0 + r) * HD + c];
    *(v8h*)&sKt[r * LDK + c] = *(const v8h*)&kp[(size_t)(col0 + r) * HD + c];
  }
  __syncthreads();

  v8f acc[4] = {};
#pragma unroll
  for (int kk = 0; kk < HD; kk += 32) {
    v16h a = frag_a(&sQ[(wv * 16) * LDQ + kk], LDQ, lane);
#pragma unroll
    for (int j = 0; j < 4; ++j) {
      v16h b = frag_bt(&sKt[(j * 16) * LDK + kk], LDK, lane);
      acc[j] = wmma32(a, b, acc[j]);
    }
  }

  const float scale = 0.125f;  // 1/sqrt(64)
  const int c0 = lane & 15, hs = lane >> 4;
#pragma unroll
  for (int j = 0; j < 4; ++j) {
#pragma unroll
    for (int r = 0; r < 8; ++r) {
      int t = row0 + wv * 16 + hs * 8 + r;
      int s = col0 + j * 16 + c0;
      Sp[(size_t)t * TT + s] = (s > t) ? -__builtin_inff() : acc[j][r] * scale;
    }
  }
}

// ---------------------------------------------------------------------------
// Row softmax over T=1024 columns, in place.  exp(-inf - max) == 0.
// ---------------------------------------------------------------------------
__global__ void __launch_bounds__(256)
softmax_kernel(float* __restrict__ A) {
  __shared__ float red[256];
  float* p = A + (size_t)blockIdx.x * TT;
  const int tid = threadIdx.x;

  float4 vv = ((const float4*)p)[tid];
  red[tid] = fmaxf(fmaxf(vv.x, vv.y), fmaxf(vv.z, vv.w));
  __syncthreads();
  for (int s = 128; s > 0; s >>= 1) {
    if (tid < s) red[tid] = fmaxf(red[tid], red[tid + s]);
    __syncthreads();
  }
  float rmax = red[0];
  __syncthreads();

  float4 e;
  e.x = __expf(vv.x - rmax);
  e.y = __expf(vv.y - rmax);
  e.z = __expf(vv.z - rmax);
  e.w = __expf(vv.w - rmax);
  red[tid] = e.x + e.y + e.z + e.w;
  __syncthreads();
  for (int s = 128; s > 0; s >>= 1) {
    if (tid < s) red[tid] += red[tid + s];
    __syncthreads();
  }
  float inv = 1.0f / red[0];
  e.x *= inv; e.y *= inv; e.z *= inv; e.w *= inv;
  ((float4*)p)[tid] = e;
}

// ---------------------------------------------------------------------------
// y0 = a @ v per (b,h): M=T, N=HD, K truncated at causal boundary.
// v tile is transposed into Bt layout [d][t-chunk] during staging.
// ---------------------------------------------------------------------------
__global__ void __launch_bounds__(128)
av_kernel(const float* __restrict__ A, const h16* __restrict__ v,
          h16* __restrict__ y0) {
  constexpr int BK = 32;
  constexpr int LDA = BK + 8;  // 40 halfs
  constexpr int LDV = BK + 8;  // sVt[d][k] : 64 x 32
  __shared__ __align__(16) h16 sA[64 * LDA];
  __shared__ __align__(16) h16 sVt[HD * LDV];

  const int tid = threadIdx.x, lane = tid & 31, wv = tid >> 5;
  const int bh = blockIdx.y;
  const int row0 = blockIdx.x * 64;
  const float* Ap = A + (size_t)bh * TT * TT;
  const h16* vp = v + (size_t)bh * TT * HD;

  v8f acc[4] = {};
  const int kend = row0 + 64;  // a[t][s]==0 for s>t after softmax
  for (int k0 = 0; k0 < kend; k0 += BK) {
    // stage A (probabilities, f32 -> f16), vectorized
    for (int idx = tid; idx < 64 * (BK / 4); idx += 128) {
      int r = idx >> 3, c = (idx & 7) * 4;
      const float4 f = *(const float4*)&Ap[(size_t)(row0 + r) * TT + k0 + c];
      v4h h = {(h16)f.x, (h16)f.y, (h16)f.z, (h16)f.w};
      *(v4h*)&sA[r * LDA + c] = h;
    }
    // stage v transposed: sVt[d][k] = v[k0+k][d]
    for (int idx = tid; idx < BK * (HD / 4); idx += 128) {
      int r = idx >> 4, c4 = (idx & 15) * 4;  // r = k, c4 = d
      v4h t = *(const v4h*)&vp[(size_t)(k0 + r) * HD + c4];
#pragma unroll
      for (int i = 0; i < 4; ++i) sVt[(c4 + i) * LDV + r] = t[i];
    }
    __syncthreads();

    v16h a = frag_a(&sA[(wv * 16) * LDA], LDA, lane);
#pragma unroll
    for (int j = 0; j < 4; ++j) {
      v16h b = frag_bt(&sVt[(j * 16) * LDV], LDV, lane);
      acc[j] = wmma32(a, b, acc[j]);
    }
    __syncthreads();
  }

  const int b = bh / HH, h = bh % HH;
  const int c0 = lane & 15, hs = lane >> 4;
#pragma unroll
  for (int j = 0; j < 4; ++j) {
#pragma unroll
    for (int r = 0; r < 8; ++r) {
      int t = row0 + wv * 16 + hs * 8 + r;
      int n = j * 16 + c0;
      y0[((size_t)b * TT + t) * CC + h * HD + n] = (h16)acc[j][r];
    }
  }
}

// ---------------------------------------------------------------------------
// Host-side orchestration
// ---------------------------------------------------------------------------
extern "C" void kernel_launch(void* const* d_in, const int* in_sizes, int n_in,
                              void* d_out, int out_size, void* d_ws, size_t ws_size,
                              hipStream_t stream) {
  (void)in_sizes; (void)n_in; (void)out_size; (void)ws_size;
  const float* Q  = (const float*)d_in[0];
  const float* K  = (const float*)d_in[1];
  const float* V  = (const float*)d_in[2];
  const float* Wq = (const float*)d_in[3];
  const float* bq = (const float*)d_in[4];
  const float* Wk = (const float*)d_in[5];
  const float* bk = (const float*)d_in[6];
  const float* Wv = (const float*)d_in[7];
  const float* bv = (const float*)d_in[8];
  const float* Wp = (const float*)d_in[9];
  const float* bp = (const float*)d_in[10];
  // d_in[11] = attn_mask: strictly-upper-triangular causal mask, hardcoded.

  float* y_out = (float*)d_out;                 // [B,T,C]
  float* a_out = y_out + (size_t)BB * TT * CC;  // [B,H,T,T]

  const size_t headElems = (size_t)BB * HH * TT * HD;  // 2M f16 each
  h16* q = (h16*)d_ws;
  h16* k = q + headElems;
  h16* v = k + headElems;
  h16* y0 = q;  // q dead after scores_kernel; reuse its 4MB for y0 [B,T,C]

  const int M = BB * TT;  // 2048
  dim3 blk(128);
  dim3 gLin(CC / 64, M / 128);  // (16, 16)

  linear_kernel<float, true><<<gLin, blk, 0, stream>>>(Q, Wq, bq, q, M, CC, CC);
  linear_kernel<float, true><<<gLin, blk, 0, stream>>>(K, Wk, bk, k, M, CC, CC);
  linear_kernel<float, true><<<gLin, blk, 0, stream>>>(V, Wv, bv, v, M, CC, CC);

  dim3 gS(TT / 64, TT / 64, BB * HH);  // (16, 16, 32)
  scores_kernel<<<gS, blk, 0, stream>>>(q, k, a_out);

  softmax_kernel<<<dim3(BB * HH * TT), dim3(256), 0, stream>>>(a_out);

  dim3 gAV(TT / 64, BB * HH);  // (16, 32)
  av_kernel<<<gAV, blk, 0, stream>>>(a_out, v, y0);

  linear_kernel<h16, false><<<gLin, blk, 0, stream>>>(y0, Wp, bp, y_out, M, CC, CC);
}
